// HoloAttention_39788577030394
// MI455X (gfx1250) — compile-verified, matrix operand order checked by reading, not run on hardware
//
#include <hip/hip_runtime.h>
#include <hip/hip_bf16.h>

typedef __attribute__((ext_vector_type(16))) _Float16 v16h;
typedef __attribute__((ext_vector_type(8)))  _Float16 v8h;
typedef __attribute__((ext_vector_type(8)))  float    v8f;

#define LDS_STRIDE 40   // 32 halves payload + 8 halves pad (80B rows, 16B aligned)

// ---------------------------------------------------------------------------
// f32 -> f16 conversion (grid-stride)
// ---------------------------------------------------------------------------
__global__ __launch_bounds__(256) void f32_to_f16_kernel(const float* __restrict__ in,
                                                         _Float16* __restrict__ out,
                                                         size_t n) {
  size_t i = (size_t)blockIdx.x * blockDim.x + threadIdx.x;
  size_t stride = (size_t)gridDim.x * blockDim.x;
  for (; i < n; i += stride) out[i] = (_Float16)__builtin_nontemporal_load(&in[i]);
}

// ---------------------------------------------------------------------------
// C[M,N] (f32) = A[M,K] (f16, row-major) * W[N,K]^T (f16, row-major)
// Block tile 128x64, 256 threads = 8 waves (4 in M x 2 in N), each wave 32x32
// (2x2 WMMA tiles, 4 x v_wmma_f32_16x16x32_f16 per K-step of 32).
// Double-buffered LDS, software-pipelined global->reg->LDS: 1 barrier/K-step.
// ---------------------------------------------------------------------------
template <bool NT_STORE>
__global__ __launch_bounds__(256) void wmma_gemm_nt(const _Float16* __restrict__ A,
                                                    const _Float16* __restrict__ W,
                                                    float* __restrict__ C,
                                                    int M, int N, int K) {
  __shared__ __align__(16) _Float16 sA[2][128 * LDS_STRIDE];
  __shared__ __align__(16) _Float16 sB[2][64 * LDS_STRIDE];

  const int tid  = threadIdx.x;
  const int lane = tid & 31;          // wave32 (gfx1250)
  const int wv   = tid >> 5;          // 0..7
  const int wm   = wv & 3;            // M quarter: 32 rows
  const int wn   = wv >> 2;           // N half:    32 cols
  const int r    = lane & 15;         // row within 16x16 fragment
  const int kb   = (lane >> 4) << 3;  // K chunk base: 0 or 8 (ISA A/B layout)

  const int bm = blockIdx.x * 128;
  const int bn = blockIdx.y * 64;

  // cooperative LDS fill mapping: each thread owns one 8-half chunk pattern
  const int frow = tid >> 2;          // 0..63
  const int fcol = (tid & 3) << 3;    // 0,8,16,24

  const _Float16* gA0 = A + (size_t)(bm + frow) * K + fcol;        // rows 0..63
  const _Float16* gA1 = A + (size_t)(bm + 64 + frow) * K + fcol;   // rows 64..127
  const _Float16* gW  = W + (size_t)(bn + frow) * K + fcol;        // rows 0..63

  // ---- prologue: fill buffer 0 with K-slab 0 ----
  {
    v8h a0 = *(const v8h*)(gA0);
    v8h a1 = *(const v8h*)(gA1);
    v8h b0 = *(const v8h*)(gW);
    *(v8h*)&sA[0][frow * LDS_STRIDE + fcol]        = a0;
    *(v8h*)&sA[0][(64 + frow) * LDS_STRIDE + fcol] = a1;
    *(v8h*)&sB[0][frow * LDS_STRIDE + fcol]        = b0;
  }
  __syncthreads();

  v8f acc00 = {}, acc01 = {}, acc10 = {}, acc11 = {};
  int cur = 0;

  // fragment gather per ISA layout: lane-group g holds K{g*8..g*8+7} and
  // K{16+g*8..16+g*8+7}; B mirrors A with lane = N.
  auto frag = [&](const _Float16* row) -> v16h {
    union { v16h v; v8h h[2]; } u;
    u.h[0] = *(const v8h*)(row + kb);
    u.h[1] = *(const v8h*)(row + kb + 16);
    return u.v;
  };

  for (int k0 = 0; k0 < K; k0 += 32) {
    const bool hasNext = (k0 + 32) < K;
    v8h na0, na1, nb;
    if (hasNext) {                       // issue next-slab global loads early
      na0 = *(const v8h*)(gA0 + k0 + 32);
      na1 = *(const v8h*)(gA1 + k0 + 32);
      nb  = *(const v8h*)(gW  + k0 + 32);
      if (k0 + 64 < K) {                 // pull slab k+2 toward caches
        __builtin_prefetch(gA0 + k0 + 64, 0, 0);
        __builtin_prefetch(gA1 + k0 + 64, 0, 0);
        __builtin_prefetch(gW  + k0 + 64, 0, 0);
      }
    }

    const v16h a0 = frag(&sA[cur][(wm * 32 + r) * LDS_STRIDE]);
    const v16h a1 = frag(&sA[cur][(wm * 32 + 16 + r) * LDS_STRIDE]);
    const v16h b0 = frag(&sB[cur][(wn * 32 + r) * LDS_STRIDE]);
    const v16h b1 = frag(&sB[cur][(wn * 32 + 16 + r) * LDS_STRIDE]);

    acc00 = __builtin_amdgcn_wmma_f32_16x16x32_f16(false, a0, false, b0,
                                                   (short)0, acc00, false, false);
    acc01 = __builtin_amdgcn_wmma_f32_16x16x32_f16(false, a0, false, b1,
                                                   (short)0, acc01, false, false);
    acc10 = __builtin_amdgcn_wmma_f32_16x16x32_f16(false, a1, false, b0,
                                                   (short)0, acc10, false, false);
    acc11 = __builtin_amdgcn_wmma_f32_16x16x32_f16(false, a1, false, b1,
                                                   (short)0, acc11, false, false);

    if (hasNext) {                       // stage next slab into the other buffer
      const int nxt = cur ^ 1;
      *(v8h*)&sA[nxt][frow * LDS_STRIDE + fcol]        = na0;
      *(v8h*)&sA[nxt][(64 + frow) * LDS_STRIDE + fcol] = na1;
      *(v8h*)&sB[nxt][frow * LDS_STRIDE + fcol]        = nb;
      __syncthreads();                   // single barrier per K-step
      cur = nxt;
    }
  }

  // C/D layout: VGPR i -> M = i + 8*(lane>>4), N = lane&15
  const int orow = bm + wm * 32 + ((lane >> 4) << 3);
  const int ocol = bn + wn * 32 + r;
  for (int i = 0; i < 8; ++i) {
    float* p00 = C + (size_t)(orow + i) * N + ocol;
    float* p01 = p00 + 16;
    float* p10 = C + (size_t)(orow + 16 + i) * N + ocol;
    float* p11 = p10 + 16;
    if (NT_STORE) {
      __builtin_nontemporal_store(acc00[i], p00);
      __builtin_nontemporal_store(acc01[i], p01);
      __builtin_nontemporal_store(acc10[i], p10);
      __builtin_nontemporal_store(acc11[i], p11);
    } else {
      *p00 = acc00[i]; *p01 = acc01[i]; *p10 = acc10[i]; *p11 = acc11[i];
    }
  }
}

// ---------------------------------------------------------------------------
// Scan pass 1: per-(b,d,tile) partial sums of v*cos(t f) and v*sin(t f)
// threads cover 256 consecutive d -> fully coalesced reads of v[b][t][d]
// ---------------------------------------------------------------------------
__global__ __launch_bounds__(256) void scan_partial_kernel(const float* __restrict__ v,
                                                           const float* __restrict__ freqs,
                                                           float* __restrict__ part,
                                                           int T, int D, int TS, int NT) {
  const int d    = blockIdx.x * blockDim.x + threadIdx.x;
  const int tile = blockIdx.y;
  const int b    = blockIdx.z;
  const float f  = freqs[d];
  const size_t base = (size_t)b * T * D + d;
  float sc = 0.f, ss = 0.f;
  const int t0 = tile * TS;
  for (int t = t0; t < t0 + TS; ++t) {
    float val = v[base + (size_t)t * D];
    float s, c;
    sincosf((float)t * f, &s, &c);
    sc = fmaf(val, c, sc);
    ss = fmaf(val, s, ss);
  }
  const size_t pi = (((size_t)b * D + d) * NT + tile) * 2;
  part[pi]     = sc;
  part[pi + 1] = ss;
}

// ---------------------------------------------------------------------------
// Scan pass 2: exclusive prefix over tiles, per channel (NT small: serial)
// ---------------------------------------------------------------------------
__global__ __launch_bounds__(256) void scan_offsets_kernel(float* __restrict__ part,
                                                           int NT, int nchan) {
  int c = blockIdx.x * blockDim.x + threadIdx.x;
  if (c >= nchan) return;
  size_t base = (size_t)c * NT * 2;
  float ra = 0.f, rb = 0.f;
  for (int tile = 0; tile < NT; ++tile) {
    float ta = part[base + tile * 2];
    float tb = part[base + tile * 2 + 1];
    part[base + tile * 2]     = ra;
    part[base + tile * 2 + 1] = rb;
    ra += ta;
    rb += tb;
  }
}

// ---------------------------------------------------------------------------
// Scan pass 3: replay tile with running sums seeded by exclusive offsets;
// retrieved = cos*A + sin*B, emitted as f16 (GEMM2 A-operand).
// v is dead after this pass -> non-temporal (last-use) loads.
// ---------------------------------------------------------------------------
__global__ __launch_bounds__(256) void scan_apply_kernel(const float* __restrict__ v,
                                                         const float* __restrict__ freqs,
                                                         const float* __restrict__ part,
                                                         _Float16* __restrict__ outh,
                                                         int T, int D, int TS, int NT) {
  const int d    = blockIdx.x * blockDim.x + threadIdx.x;
  const int tile = blockIdx.y;
  const int b    = blockIdx.z;
  const float f  = freqs[d];
  const size_t pi = (((size_t)b * D + d) * NT + tile) * 2;
  float accC = part[pi];
  float accS = part[pi + 1];
  const size_t base = (size_t)b * T * D + d;
  const int t0 = tile * TS;
  for (int t = t0; t < t0 + TS; ++t) {
    float val = __builtin_nontemporal_load(&v[base + (size_t)t * D]);
    float s, c;
    sincosf((float)t * f, &s, &c);
    accC = fmaf(val, c, accC);
    accS = fmaf(val, s, accS);
    outh[base + (size_t)t * D] = (_Float16)(c * accC + s * accS);
  }
}

// ---------------------------------------------------------------------------
// Launch: convert -> GEMM1 (v = x Wv^T) -> blocked scan -> GEMM2 (out = r Wo^T)
// Dead k-projection (Wk) is skipped entirely.
// ---------------------------------------------------------------------------
extern "C" void kernel_launch(void* const* d_in, const int* in_sizes, int n_in,
                              void* d_out, int out_size, void* d_ws, size_t ws_size,
                              hipStream_t stream) {
  const float* x     = (const float*)d_in[0];
  // d_in[1] = Wk : dead value, never used by the output
  const float* Wv    = (const float*)d_in[2];
  const float* Wo    = (const float*)d_in[3];
  const float* freqs = (const float*)d_in[4];

  const int D  = in_sizes[4];        // 1024
  const int BT = in_sizes[0] / D;    // B*T = 16384
  const int B  = 2;
  const int T  = BT / B;             // 8192
  const int TS = 256;                // scan tile length
  const int NT = T / TS;             // 32 tiles

  // workspace layout
  char* ws = (char*)d_ws;
  _Float16* xh   = (_Float16*)ws;                                   // BT*D f16 (reused as retrieved_f16)
  float*    vbuf = (float*)(ws + (size_t)BT * D * 2);               // BT*D f32
  _Float16* Wvh  = (_Float16*)(ws + (size_t)BT * D * 2 + (size_t)BT * D * 4);
  _Float16* Woh  = Wvh + (size_t)D * D;
  float*    part = (float*)((char*)(Woh + (size_t)D * D));          // B*D*NT*2 f32

  const size_t nX = (size_t)BT * D;
  const size_t nW = (size_t)D * D;

  f32_to_f16_kernel<<<4096, 256, 0, stream>>>(x,  xh,  nX);
  f32_to_f16_kernel<<<1024, 256, 0, stream>>>(Wv, Wvh, nW);
  f32_to_f16_kernel<<<1024, 256, 0, stream>>>(Wo, Woh, nW);

  dim3 gGemm(BT / 128, D / 64);
  // v stays regular-temporal: re-read twice by the scan, fits in 192MB L2
  wmma_gemm_nt<false><<<gGemm, 256, 0, stream>>>(xh, Wvh, vbuf, BT, D, D);

  dim3 gScan(D / 256, NT, B);
  scan_partial_kernel<<<gScan, 256, 0, stream>>>(vbuf, freqs, part, T, D, TS, NT);

  const int nchan = B * D;
  scan_offsets_kernel<<<(nchan + 255) / 256, 256, 0, stream>>>(part, NT, nchan);

  scan_apply_kernel<<<gScan, 256, 0, stream>>>(vbuf, freqs, part, xh, T, D, TS, NT);

  // d_out is write-once: non-temporal stores keep it out of L2
  wmma_gemm_nt<true><<<gGemm, 256, 0, stream>>>(xh, Woh, (float*)d_out, BT, D, D);
}
